// MultiHeadRecurrentActorNetwork_22557168239499
// MI455X (gfx1250) — compile-verified, matrix operand order checked by reading, not run on hardware
//
#include <hip/hip_runtime.h>
#include <hip/hip_bf16.h>
#include <math.h>

typedef __attribute__((ext_vector_type(16))) _Float16 v16h;
typedef __attribute__((ext_vector_type(8)))  float    v8f;

#define NEGV (-1e8f)
#define WAVES 4
#define ROWS_PER_WAVE 16
#define ROWS_PER_BLOCK (WAVES * ROWS_PER_WAVE)
#define NTHREADS 128

__device__ __forceinline__ int kmap(int v, int grp) {
  // 16-bit A/B fragment K packing per CDNA5 ISA 7.12.2:
  // VGPR v<4 -> K = 2v (+8 for lanes 16..31); v>=4 -> K = 16+2(v-4) (+8)
  return (v < 4 ? 2 * v : 16 + 2 * (v - 4)) + 8 * grp;
}

__device__ __forceinline__ v8f wmma_f16(v16h a, v16h b, v8f c) {
  return __builtin_amdgcn_wmma_f32_16x16x32_f16(false, a, false, b, (short)0, c,
                                                false, false);
}

// ---- LDS weight staging: KxN f32 weight -> f16 fragments in WMMA-B order.
// Fragment (kt,nt) occupies 512 halves: lane l's 16 halves are contiguous.
__device__ __forceinline__ void stage_weight(_Float16* dst,
                                             const float* __restrict__ W,
                                             int K, int N, int tid) {
  const int NT = N >> 4;
  const int total = K * N;
  for (int idx = tid; idx < total; idx += NTHREADS) {
    int e = idx & 15;
    int l = (idx >> 4) & 31;
    int frag = idx >> 9;
    int kt = frag / NT, nt = frag - kt * NT;
    int k = kt * 32 + kmap(e >> 1, l >> 4) + (e & 1);
    int n = nt * 16 + (l & 15);
    dst[idx] = (_Float16)W[k * N + n];
  }
}

// Head weight 256x16 (cols: 0-1 pick, 2-3 partner, 4 pu, 5-15 zero), NT = 1.
__device__ __forceinline__ void stage_head(_Float16* dst,
                                           const float* __restrict__ Wp,
                                           const float* __restrict__ Wpa,
                                           const float* __restrict__ Wpu,
                                           int tid) {
  for (int idx = tid; idx < 256 * 16; idx += NTHREADS) {
    int e = idx & 15;
    int l = (idx >> 4) & 31;
    int kt = idx >> 9;
    int k = kt * 32 + kmap(e >> 1, l >> 4) + (e & 1);
    int n = l & 15;
    float val = 0.f;
    if (n < 2)       val = Wp[k * 2 + n];
    else if (n < 4)  val = Wpa[k * 2 + (n - 2)];
    else if (n == 4) val = Wpu[k];
    dst[idx] = (_Float16)val;
  }
}

// B fragment from staged LDS: two ds_load_b128 (32 contiguous bytes).
__device__ __forceinline__ v16h load_b_frag(const _Float16* sW, int NT, int kt,
                                            int nt, int lane) {
  return *(const v16h*)(sW + ((((kt * NT + nt) << 5) + lane) << 4));
}

// A fragment: 16x32 f16 tile from a row-major f32 matrix (float2 loads).
__device__ __forceinline__ v16h load_a_f32(const float* __restrict__ p0,
                                           int lane) {
  const float2* p2 = (const float2*)(p0 + 8 * (lane >> 4));
  v16h a;
#pragma unroll
  for (int v = 0; v < 8; ++v) {
    int o2 = (v < 4) ? v : v + 4;  // float2 offsets {0,1,2,3, 8,9,10,11}
    float2 x = p2[o2];
    a[2 * v]     = (_Float16)x.x;
    a[2 * v + 1] = (_Float16)x.y;
  }
  return a;
}

__global__ __launch_bounds__(NTHREADS) void actor_fused_kernel(
    const float* __restrict__ features, const float* __restrict__ hand_tokens,
    const int* __restrict__ hand_ids, const unsigned char* __restrict__ action_mask,
    const float* __restrict__ card_table,
    const float* __restrict__ W_pick, const float* __restrict__ b_pick,
    const float* __restrict__ W_partner, const float* __restrict__ b_partner,
    const float* __restrict__ W_pu, const float* __restrict__ b_pu,
    const float* __restrict__ Wg_ptr, const float* __restrict__ bg_ptr,
    const float* __restrict__ Wt_ptr, const float* __restrict__ bt_ptr,
    const float* __restrict__ v_ptr,
    const float* __restrict__ Wg_tw, const float* __restrict__ bg_tw,
    const float* __restrict__ We_tw, const float* __restrict__ be_tw,
    float* __restrict__ out) {
  // Staged f16 weights in WMMA-B fragment layout (80 KB)
  __shared__ __align__(32) _Float16 sWgp[256 * 64];  // 8 kt x 4 nt
  __shared__ __align__(32) _Float16 sWtw[256 * 64];  // 8 kt x 4 nt
  __shared__ __align__(32) _Float16 sWt[64 * 64];    // 2 kt x 4 nt
  __shared__ __align__(32) _Float16 sWh[256 * 16];   // 8 kt x 1 nt
  // Per-wave scratch
  __shared__ _Float16 sGP[WAVES][16 * 64];  // g_ptr (f16)
  __shared__ _Float16 sGT[WAVES][16 * 64];  // g_tw  (f16)
  __shared__ float sSS[WAVES][16 * 8];      // slot scores
  __shared__ float sHead[WAVES][16 * 8];    // pick0,pick1,par0,par1,pu
  __shared__ float sTW6[WAVES][16 * 8];     // two-tower scores (6 call cards)
  __shared__ int   sIds[WAVES][16 * 8];     // hand ids
  __shared__ float sE6[6 * 64];             // card_table[cids]@We_tw + be_tw

  const int tid  = threadIdx.x;
  const int lane = tid & 31;
  const int w    = tid >> 5;
  const int grp  = lane >> 4;
  const int n    = lane & 15;
  const int R0   = blockIdx.x * ROWS_PER_BLOCK + w * ROWS_PER_WAVE;

  // ---- Phase 1: stage weights (f32->f16, fragment-swizzled), E6, ids
  stage_weight(sWgp, Wg_ptr, 256, 64, tid);
  stage_weight(sWtw, Wg_tw, 256, 64, tid);
  stage_weight(sWt, Wt_ptr, 64, 64, tid);
  stage_head(sWh, W_pick, W_partner, W_pu, tid);
  {
    const int cids[6] = {0, 2, 4, 6, 8, 10};
    for (int i = tid; i < 6 * 64; i += NTHREADS) {
      int c = i >> 6, col = i & 63;
      float s = be_tw[col];
      const float* ct = card_table + cids[c] * 64;
#pragma unroll 8
      for (int k = 0; k < 64; ++k) s += ct[k] * We_tw[k * 64 + col];
      sE6[i] = s;
    }
  }
#pragma unroll
  for (int j = 0; j < 4; ++j) {
    int i = lane + 32 * j;  // 0..127 = 16 rows x 8 slots
    sIds[w][i] = hand_ids[(size_t)(R0 + (i >> 3)) * 8 + (i & 7)];
  }
  __syncthreads();

  // ---- Phase 2: GEMM1  X(16x256) @ {Whead | Wg_ptr | Wg_tw}
  v16h af[8];
#pragma unroll
  for (int kt = 0; kt < 8; ++kt)
    af[kt] = load_a_f32(features + (size_t)(R0 + n) * 256 + kt * 32, lane);

  {  // head tile (5 live columns)
    v8f d = {};
#pragma unroll
    for (int kt = 0; kt < 8; ++kt)
      d = wmma_f16(af[kt], load_b_frag(sWh, 1, kt, 0, lane), d);
    if (n < 5) {
      float bb = (n < 2) ? b_pick[n] : ((n < 4) ? b_partner[n - 2] : b_pu[0]);
#pragma unroll
      for (int v = 0; v < 8; ++v) sHead[w][(v + 8 * grp) * 8 + n] = d[v] + bb;
    }
  }
#pragma unroll
  for (int nt = 0; nt < 4; ++nt) {  // pointer g
    v8f d = {};
#pragma unroll
    for (int kt = 0; kt < 8; ++kt)
      d = wmma_f16(af[kt], load_b_frag(sWgp, 4, kt, nt, lane), d);
    int col = nt * 16 + n;
    float bb = bg_ptr[col];
#pragma unroll
    for (int v = 0; v < 8; ++v)
      sGP[w][(v + 8 * grp) * 64 + col] = (_Float16)(d[v] + bb);
  }
#pragma unroll
  for (int nt = 0; nt < 4; ++nt) {  // two-tower g
    v8f d = {};
#pragma unroll
    for (int kt = 0; kt < 8; ++kt)
      d = wmma_f16(af[kt], load_b_frag(sWtw, 4, kt, nt, lane), d);
    int col = nt * 16 + n;
    float bb = bg_tw[col];
#pragma unroll
    for (int v = 0; v < 8; ++v)
      sGT[w][(v + 8 * grp) * 64 + col] = (_Float16)(d[v] + bb);
  }
  __syncthreads();

  // ---- Phase 3: token GEMM (128 rs-rows x 64) @ Wt, fused tanh + v_ptr dot
  for (int mt = 0; mt < 8; ++mt) {
    int rsA = mt * 16 + n;  // A-row for this lane
    const float* tp =
        hand_tokens + ((size_t)(R0 + (rsA >> 3)) * 8 + (rsA & 7)) * 64;
    v16h ta0 = load_a_f32(tp, lane);
    v16h ta1 = load_a_f32(tp + 32, lane);
    float partial[8] = {0.f, 0.f, 0.f, 0.f, 0.f, 0.f, 0.f, 0.f};
#pragma unroll
    for (int nt = 0; nt < 4; ++nt) {
      v8f d = {};
      d = wmma_f16(ta0, load_b_frag(sWt, 4, 0, nt, lane), d);
      d = wmma_f16(ta1, load_b_frag(sWt, 4, 1, nt, lane), d);
      int col = nt * 16 + n;
      float vp = v_ptr[col], bt = bt_ptr[col];
#pragma unroll
      for (int v = 0; v < 8; ++v) {
        int rs = mt * 16 + v + 8 * grp;
        float g = (float)sGP[w][(rs >> 3) * 64 + col];
        partial[v] += tanhf(d[v] + g + bt) * vp;
      }
    }
    // reduce each D-row across its 16-lane half (xor<16 stays within half)
#pragma unroll
    for (int v = 0; v < 8; ++v) {
      float x = partial[v];
      x += __shfl_xor(x, 1);
      x += __shfl_xor(x, 2);
      x += __shfl_xor(x, 4);
      x += __shfl_xor(x, 8);
      if (n == 0) {
        int rs = mt * 16 + v + 8 * grp;  // rs == r*8+s flat
        sSS[w][rs] = x;
      }
    }
  }
  // two-tower scores: 16 rows x 6 call cards, dot-64 each (3 per lane)
#pragma unroll
  for (int j = 0; j < 3; ++j) {
    int i = lane * 3 + j;  // 0..95
    int r = i / 6, c = i % 6;
    float s = 0.f;
#pragma unroll 8
    for (int k = 0; k < 64; ++k)
      s += (float)sGT[w][r * 64 + k] * sE6[c * 64 + k];
    sTW6[w][r * 8 + c] = s;
  }
  __syncthreads();

  // ---- Phase 4: assemble 107 logits per row, mask, softmax, store
  for (int r = 0; r < 16; ++r) {
    const int grow = R0 + r;
    float vals[4];
#pragma unroll
    for (int j = 0; j < 4; ++j) {
      int a = lane + 32 * j;
      float lg = NEGV;
      if (a < 107) {
        if (a < 4) {
          lg = sHead[w][r * 8 + a];  // pick0,pick1,alone,jd
        } else if (a < 10) {
          lg = sTW6[w][r * 8 + (a - 4)];  // call actions
        } else if (a < 106) {
          int cid = (a - 10) & 31;  // bury/under/play share cid
#pragma unroll
          for (int s = 0; s < 8; ++s)
            if (sIds[w][r * 8 + s] == cid) lg = sSS[w][r * 8 + s];
        } else {
          lg = sHead[w][r * 8 + 4];  // play-under
        }
        if (!action_mask[(size_t)grow * 107 + a]) lg = NEGV;
      }
      vals[j] = lg;
    }
    float mx = fmaxf(fmaxf(vals[0], vals[1]), fmaxf(vals[2], vals[3]));
    mx = fmaxf(mx, __shfl_xor(mx, 1));
    mx = fmaxf(mx, __shfl_xor(mx, 2));
    mx = fmaxf(mx, __shfl_xor(mx, 4));
    mx = fmaxf(mx, __shfl_xor(mx, 8));
    mx = fmaxf(mx, __shfl_xor(mx, 16));
    float e[4], se = 0.f;
#pragma unroll
    for (int j = 0; j < 4; ++j) {
      int a = lane + 32 * j;
      e[j] = (a < 107) ? __expf(vals[j] - mx) : 0.f;
      se += e[j];
    }
    se += __shfl_xor(se, 1);
    se += __shfl_xor(se, 2);
    se += __shfl_xor(se, 4);
    se += __shfl_xor(se, 8);
    se += __shfl_xor(se, 16);
    float inv = 1.f / se;
#pragma unroll
    for (int j = 0; j < 4; ++j) {
      int a = lane + 32 * j;
      if (a < 107) out[(size_t)grow * 107 + a] = e[j] * inv;
    }
  }
}

extern "C" void kernel_launch(void* const* d_in, const int* in_sizes, int n_in,
                              void* d_out, int out_size, void* d_ws, size_t ws_size,
                              hipStream_t stream) {
  const float* features          = (const float*)d_in[0];
  const float* hand_tokens       = (const float*)d_in[1];
  const int* hand_ids            = (const int*)d_in[2];
  const unsigned char* action_mask = (const unsigned char*)d_in[3];
  const float* card_table        = (const float*)d_in[4];
  const float* W_pick    = (const float*)d_in[5];
  const float* b_pick    = (const float*)d_in[6];
  const float* W_partner = (const float*)d_in[7];
  const float* b_partner = (const float*)d_in[8];
  const float* W_pu      = (const float*)d_in[9];
  const float* b_pu      = (const float*)d_in[10];
  const float* Wg_ptr    = (const float*)d_in[11];
  const float* bg_ptr    = (const float*)d_in[12];
  const float* Wt_ptr    = (const float*)d_in[13];
  const float* bt_ptr    = (const float*)d_in[14];
  const float* v_ptr     = (const float*)d_in[15];
  const float* Wg_tw     = (const float*)d_in[16];
  const float* bg_tw     = (const float*)d_in[17];
  const float* We_tw     = (const float*)d_in[18];
  const float* be_tw     = (const float*)d_in[19];
  float* out = (float*)d_out;

  const int B = in_sizes[0] / 256;      // 131072
  const int grid = B / ROWS_PER_BLOCK;  // 2048 blocks of 128 threads (4 waves)
  actor_fused_kernel<<<grid, NTHREADS, 0, stream>>>(
      features, hand_tokens, hand_ids, action_mask, card_table,
      W_pick, b_pick, W_partner, b_partner, W_pu, b_pu,
      Wg_ptr, bg_ptr, Wt_ptr, bt_ptr, v_ptr,
      Wg_tw, bg_tw, We_tw, be_tw, out);
  (void)n_in; (void)out_size; (void)d_ws; (void)ws_size;
}
